// GAT_54279796687493
// MI455X (gfx1250) — compile-verified, matrix operand order checked by reading, not run on hardware
//
#include <hip/hip_runtime.h>
#include <hip/hip_bf16.h>
#include <math.h>

typedef __attribute__((ext_vector_type(16))) _Float16 v16h;
typedef __attribute__((ext_vector_type(8)))  float    v8f;

#define WAVES 2

// K index inside a 16x32 f16 WMMA operand fragment for (lane, element e=0..15),
// per CDNA5 ISA 7.12.2: VGPR v holds K pairs; v<4 -> K=2v+p, v>=4 -> K=16+2(v-4)+p,
// +8 for lanes 16..31.
__device__ __forceinline__ int kfrag(int lane, int e) {
    int v = e >> 1, p = e & 1;
    return ((v & 4) << 2) + ((v & 3) << 1) + p + ((lane >> 4) << 3);
}

__device__ __forceinline__ v8f wmma16(v16h a, v16h b, v8f c) {
    return __builtin_amdgcn_wmma_f32_16x16x32_f16(false, a, false, b, (short)0, c,
                                                  false, false);
}

__device__ __forceinline__ float sigf(float x) { return 1.0f / (1.0f + expf(-x)); }

// ---------------------------------------------------------------------------
// Pack an f32 weight matrix into per-(ktile,jtile) WMMA B-fragment layout, f16.
// transposed==1: B[k][n] = src[n*K + k]   (the  x @ W.T  case, W is [N,K])
// transposed==0: B[k][n] = src[k*N + n]   (the  x @ W    case, W is [K,N])
// dst tile (kt,jt) lives at (jt*numKt+kt)*512 halfs; lane l owns halfs [l*16, l*16+16).
// ---------------------------------------------------------------------------
__global__ void pack_b_kernel(const float* __restrict__ src, _Float16* __restrict__ dst,
                              int K, int Ncols, int transposed) {
    int t = blockIdx.x * blockDim.x + threadIdx.x;
    int total = K * Ncols;
    if (t >= total) return;
    int tile = t >> 9, within = t & 511;
    int lane = within >> 4, e = within & 15;
    int numKt = K >> 5;
    int kt = tile % numKt, jt = tile / numKt;
    int k = kt * 32 + kfrag(lane, e);
    int n = jt * 16 + (lane & 15);
    float v = transposed ? src[(size_t)n * K + k] : src[(size_t)k * Ncols + n];
    dst[t] = (_Float16)v;
}

// ---------------------------------------------------------------------------
// Fused SAGEConv(lstm): per wave, 16 nodes.  8-step LSTM over gathered
// neighbors with WMMA gate GEMMs (bias in C), fp32 gate math, c-state in VGPRs,
// h-state in LDS; then out = self @ Wself + agg @ Wneigh + b (WMMA), opt. relu.
// ---------------------------------------------------------------------------
template <int D>
__global__ void __launch_bounds__(WAVES * 32)
sage_lstm_kernel(const float* __restrict__ xin, const int* __restrict__ adj,
                 const _Float16* __restrict__ Bih, const _Float16* __restrict__ Bhh,
                 const float* __restrict__ blstm,
                 const _Float16* __restrict__ Bself, const _Float16* __restrict__ Bneigh,
                 const float* __restrict__ bout,
                 float* __restrict__ out, int relu_flag, int N) {
    constexpr int KT  = D / 32;   // K tiles per GEMM (2 for D=64, 1 for D=32)
    constexpr int DT  = D / 16;   // hidden-dim tiles (4 or 2)
    constexpr int PAD = 8;
    __shared__ float xs[WAVES][16][D + PAD];
    __shared__ float hs[WAVES][16][D + PAD];

    const int w    = threadIdx.x >> 5;
    const int lane = threadIdx.x & 31;
    const int row  = lane & 15;
    const int hi   = lane >> 4;
    const int n0   = (blockIdx.x * WAVES + w) * 16;
    if (n0 >= N) return;

    for (int i = lane; i < 16 * D; i += 32) hs[w][i / D][i % D] = 0.0f;

    float cst[DT][8];
#pragma unroll
    for (int dt = 0; dt < DT; ++dt)
#pragma unroll
        for (int r = 0; r < 8; ++r) cst[dt][r] = 0.0f;

    __syncthreads();

    for (int t = 0; t < 8; ++t) {
        // gather x_t = xin[adj[n, t]] (16 rows x D) into LDS, float4 chunks
        for (int i = lane; i < 16 * (D / 4); i += 32) {
            int r = i / (D / 4), c4 = i % (D / 4);
            int s = adj[(size_t)(n0 + r) * 8 + t];
            *((float4*)&xs[w][r][c4 * 4]) =
                ((const float4*)(xin + (size_t)s * D))[c4];
        }
        __syncthreads();

        // build A fragments for x_t and h (f32 -> f16)
        v16h ax[KT], ah[KT];
#pragma unroll
        for (int kt = 0; kt < KT; ++kt)
#pragma unroll
            for (int e = 0; e < 16; ++e) {
                int k = kt * 32 + kfrag(lane, e);
                ax[kt][e] = (_Float16)xs[w][row][k];
                ah[kt][e] = (_Float16)hs[w][row][k];
            }
        __syncthreads();  // fragments captured; hs may now be overwritten

#pragma unroll
        for (int dt = 0; dt < DT; ++dt) {
            v8f acc[4];
#pragma unroll
            for (int g = 0; g < 4; ++g) {       // gate order i,f,g,o
                int jt = g * DT + dt;
                float bv = blstm[jt * 16 + row];
                v8f c;
#pragma unroll
                for (int r = 0; r < 8; ++r) c[r] = bv;
#pragma unroll
                for (int kt = 0; kt < KT; ++kt) {
                    v16h bw = *(const v16h*)(Bih + (size_t)(jt * KT + kt) * 512 + lane * 16);
                    c = wmma16(ax[kt], bw, c);
                    v16h bh = *(const v16h*)(Bhh + (size_t)(jt * KT + kt) * 512 + lane * 16);
                    c = wmma16(ah[kt], bh, c);
                }
                acc[g] = c;
            }
#pragma unroll
            for (int r = 0; r < 8; ++r) {
                float ig = sigf(acc[0][r]);
                float fg = sigf(acc[1][r]);
                float gg = tanhf(acc[2][r]);
                float og = sigf(acc[3][r]);
                float cc = fg * cst[dt][r] + ig * gg;
                cst[dt][r] = cc;
                hs[w][r + 8 * hi][dt * 16 + row] = og * tanhf(cc);
            }
        }
        __syncthreads();
    }

    // out = self @ Wself + agg @ Wneigh + b  (agg == hs)
    for (int i = lane; i < 16 * (D / 4); i += 32) {
        int r = i / (D / 4), c4 = i % (D / 4);
        *((float4*)&xs[w][r][c4 * 4]) =
            ((const float4*)(xin + (size_t)(n0 + r) * D))[c4];
    }
    __syncthreads();

    v16h as[KT], ag[KT];
#pragma unroll
    for (int kt = 0; kt < KT; ++kt)
#pragma unroll
        for (int e = 0; e < 16; ++e) {
            int k = kt * 32 + kfrag(lane, e);
            as[kt][e] = (_Float16)xs[w][row][k];
            ag[kt][e] = (_Float16)hs[w][row][k];
        }

#pragma unroll
    for (int jt = 0; jt < 2; ++jt) {           // OUT = 32 -> 2 column tiles
        float bv = bout[jt * 16 + row];
        v8f c;
#pragma unroll
        for (int r = 0; r < 8; ++r) c[r] = bv;
#pragma unroll
        for (int kt = 0; kt < KT; ++kt) {
            v16h b1 = *(const v16h*)(Bself + (size_t)(jt * KT + kt) * 512 + lane * 16);
            c = wmma16(as[kt], b1, c);
            v16h b2 = *(const v16h*)(Bneigh + (size_t)(jt * KT + kt) * 512 + lane * 16);
            c = wmma16(ag[kt], b2, c);
        }
#pragma unroll
        for (int r = 0; r < 8; ++r) {
            float v = c[r];
            if (relu_flag) v = fmaxf(v, 0.0f);
            out[(size_t)(n0 + r + 8 * hi) * 32 + jt * 16 + row] = v;
        }
    }
}

// ---------------------------------------------------------------------------
// NNConv: m[e] = h_src[e] @ (efeat[e] @ edge_W + edge_b).reshape(32,32)
//       = [A2 row] @ B2,  A2[e, k*32+i] = efeat[e,k]*h[e,i], B2 = edge_W [512,32]
// plus h_src[e] @ edge_b[32,32].  Then sum over deg=8, + nn_b, relu.
// One wave = 16 nodes = 8 edge-tiles of 16 edges.
// ---------------------------------------------------------------------------
__global__ void __launch_bounds__(WAVES * 32)
nnconv_kernel(const float* __restrict__ hin, const int* __restrict__ adj,
              const float* __restrict__ efeat,
              const _Float16* __restrict__ BeW,  // packed [512][32], numKt=16
              const _Float16* __restrict__ BeB,  // packed [32][32],  numKt=1
              const float* __restrict__ nnb,
              float* __restrict__ out, int N) {
    __shared__ float hsrc[WAVES][16][36];
    __shared__ float ef[WAVES][16][16];
    const int w    = threadIdx.x >> 5;
    const int lane = threadIdx.x & 31;
    const int row  = lane & 15;
    const int hi   = lane >> 4;
    const int n0   = (blockIdx.x * WAVES + w) * 16;
    if (n0 >= N) return;

    for (int et = 0; et < 8; ++et) {
        const int ebase = n0 * 8 + et * 16;
        // stage 16 source rows (32 f32) and 16 efeat rows (16 f32)
        for (int i = lane; i < 16 * 8; i += 32) {
            int le = i >> 3, c4 = i & 7;
            int s = adj[ebase + le];
            *((float4*)&hsrc[w][le][c4 * 4]) =
                ((const float4*)(hin + (size_t)s * 32))[c4];
        }
        for (int i = lane; i < 16 * 4; i += 32) {
            int le = i >> 2, c4 = i & 3;
            *((float4*)&ef[w][le][c4 * 4]) =
                ((const float4*)(efeat + (size_t)(ebase + le) * 16))[c4];
        }
        __syncthreads();

        float hr[16];
#pragma unroll
        for (int e = 0; e < 16; ++e) hr[e] = hsrc[w][row][kfrag(lane, e)];

        v8f c0, c1;
#pragma unroll
        for (int r = 0; r < 8; ++r) { c0[r] = 0.0f; c1[r] = 0.0f; }

        for (int kk = 0; kk < 16; ++kk) {
            float sc = ef[w][row][kk];
            v16h a;
#pragma unroll
            for (int e = 0; e < 16; ++e) a[e] = (_Float16)(sc * hr[e]);
            v16h b0 = *(const v16h*)(BeW + (size_t)(0 * 16 + kk) * 512 + lane * 16);
            c0 = wmma16(a, b0, c0);
            v16h b1 = *(const v16h*)(BeW + (size_t)(1 * 16 + kk) * 512 + lane * 16);
            c1 = wmma16(a, b1, c1);
        }
        {   // edge_b contribution: h_src @ edge_b[32,32]
            v16h a;
#pragma unroll
            for (int e = 0; e < 16; ++e) a[e] = (_Float16)hr[e];
            v16h b0 = *(const v16h*)(BeB + (size_t)0 * 512 + lane * 16);
            c0 = wmma16(a, b0, c0);
            v16h b1 = *(const v16h*)(BeB + (size_t)1 * 512 + lane * 16);
            c1 = wmma16(a, b1, c1);
        }
        // rows 0..7 of the C tile are the 8 deg slots of node n0+2*et (+hi)
        float s0 = 0.0f, s1 = 0.0f;
#pragma unroll
        for (int r = 0; r < 8; ++r) { s0 += c0[r]; s1 += c1[r]; }
        int n = n0 + 2 * et + hi;
        out[(size_t)n * 32 + row]      = fmaxf(s0 + nnb[row], 0.0f);
        out[(size_t)n * 32 + 16 + row] = fmaxf(s1 + nnb[16 + row], 0.0f);
        __syncthreads();
    }
}

// ---------------------------------------------------------------------------
// Global attention pooling per graph (ng nodes/graph), softmax gate, elu.
// ---------------------------------------------------------------------------
__global__ void __launch_bounds__(256)
pool_kernel(const float* __restrict__ h, const float* __restrict__ gW1,
            const float* __restrict__ gb1, const float* __restrict__ gW2,
            const float* __restrict__ gb2, float* __restrict__ r) {
    __shared__ float red[256];
    __shared__ float acc[32];
    const int tid = threadIdx.x;
    const int n = blockIdx.x * blockDim.x + tid;

    float hv[32];
#pragma unroll
    for (int d = 0; d < 32; ++d) hv[d] = h[(size_t)n * 32 + d];

    float gate = gb2[0];
#pragma unroll
    for (int j = 0; j < 16; ++j) {
        float z = gb1[j];
#pragma unroll
        for (int d = 0; d < 32; ++d) z += hv[d] * gW1[d * 16 + j];
        gate += fmaxf(z, 0.0f) * gW2[j];
    }

    red[tid] = gate;
    __syncthreads();
    for (int s = blockDim.x >> 1; s > 0; s >>= 1) {
        if (tid < s) red[tid] = fmaxf(red[tid], red[tid + s]);
        __syncthreads();
    }
    float m = red[0];
    __syncthreads();
    float ex = expf(gate - m);
    red[tid] = ex;
    __syncthreads();
    for (int s = blockDim.x >> 1; s > 0; s >>= 1) {
        if (tid < s) red[tid] += red[tid + s];
        __syncthreads();
    }
    float wgt = ex / red[0];

    if (tid < 32) acc[tid] = 0.0f;
    __syncthreads();
#pragma unroll
    for (int d = 0; d < 32; ++d) atomicAdd(&acc[d], wgt * hv[d]);
    __syncthreads();
    if (tid < 32) {
        float v = acc[tid];
        v = v > 0.0f ? v : (expf(v) - 1.0f);  // elu
        r[blockIdx.x * 32 + tid] = v;
    }
}

__global__ void __launch_bounds__(64)
final_mlp_kernel(const float* __restrict__ r,
                 const float* __restrict__ W1, const float* __restrict__ b1,
                 const float* __restrict__ W2, const float* __restrict__ b2,
                 const float* __restrict__ W3, const float* __restrict__ b3,
                 float* __restrict__ out) {
    int g = threadIdx.x;
    float v[32], z[32];
#pragma unroll
    for (int d = 0; d < 32; ++d) v[d] = r[g * 32 + d];
    for (int j = 0; j < 32; ++j) {
        float s = b1[j];
#pragma unroll
        for (int d = 0; d < 32; ++d) s += v[d] * W1[d * 32 + j];
        z[j] = fmaxf(s, 0.0f);
    }
    for (int j = 0; j < 32; ++j) {
        float s = b2[j];
#pragma unroll
        for (int d = 0; d < 32; ++d) s += z[d] * W2[d * 32 + j];
        v[j] = fmaxf(s, 0.0f);
    }
    float o = b3[0];
#pragma unroll
    for (int d = 0; d < 32; ++d) o += v[d] * W3[d];
    out[g] = o;
}

// ---------------------------------------------------------------------------
extern "C" void kernel_launch(void* const* d_in, const int* in_sizes, int n_in,
                              void* d_out, int out_size, void* d_ws, size_t ws_size,
                              hipStream_t stream) {
    const float* x        = (const float*)d_in[0];
    const int*   adj      = (const int*)d_in[1];
    const float* efeat    = (const float*)d_in[2];
    // d_in[3] = num_graphs (fixed 64 for this problem)
    const float* l1_Wih   = (const float*)d_in[4];
    const float* l1_Whh   = (const float*)d_in[5];
    const float* l1_b     = (const float*)d_in[6];
    const float* s1_Wself = (const float*)d_in[7];
    const float* s1_Wnb   = (const float*)d_in[8];
    const float* s1_b     = (const float*)d_in[9];
    const float* l2_Wih   = (const float*)d_in[10];
    const float* l2_Whh   = (const float*)d_in[11];
    const float* l2_b     = (const float*)d_in[12];
    const float* s2_Wself = (const float*)d_in[13];
    const float* s2_Wnb   = (const float*)d_in[14];
    const float* s2_b     = (const float*)d_in[15];
    const float* edge_W   = (const float*)d_in[16];
    const float* edge_b   = (const float*)d_in[17];
    const float* nn_b     = (const float*)d_in[18];
    const float* gate_W1  = (const float*)d_in[19];
    const float* gate_b1  = (const float*)d_in[20];
    const float* gate_W2  = (const float*)d_in[21];
    const float* gate_b2  = (const float*)d_in[22];
    const float* fin_W1   = (const float*)d_in[23];
    const float* fin_b1   = (const float*)d_in[24];
    const float* fin_W2   = (const float*)d_in[25];
    const float* fin_b2   = (const float*)d_in[26];
    const float* fin_W3   = (const float*)d_in[27];
    const float* fin_b3   = (const float*)d_in[28];

    const int N  = in_sizes[0] / 64;   // 16384
    const int NG = 64;                 // num_graphs (setup-fixed)

    // ---- workspace carve (256B aligned) ----
    char* wsb = (char*)d_ws;
    auto carve = [&](size_t bytes) -> void* {
        void* p = (void*)wsb;
        wsb += (bytes + 255) & ~(size_t)255;
        return p;
    };
    float* h1 = (float*)carve((size_t)N * 32 * sizeof(float));
    float* h2 = (float*)carve((size_t)N * 32 * sizeof(float));
    float* rp = (float*)carve((size_t)NG * 32 * sizeof(float));
    _Float16* pL1ih = (_Float16*)carve(64 * 256 * sizeof(_Float16));
    _Float16* pL1hh = (_Float16*)carve(64 * 256 * sizeof(_Float16));
    _Float16* pS1s  = (_Float16*)carve(64 * 32 * sizeof(_Float16));
    _Float16* pS1n  = (_Float16*)carve(64 * 32 * sizeof(_Float16));
    _Float16* pL2ih = (_Float16*)carve(32 * 128 * sizeof(_Float16));
    _Float16* pL2hh = (_Float16*)carve(32 * 128 * sizeof(_Float16));
    _Float16* pS2s  = (_Float16*)carve(32 * 32 * sizeof(_Float16));
    _Float16* pS2n  = (_Float16*)carve(32 * 32 * sizeof(_Float16));
    _Float16* pEW   = (_Float16*)carve(512 * 32 * sizeof(_Float16));
    _Float16* pEB   = (_Float16*)carve(32 * 32 * sizeof(_Float16));
    float* h3 = h1;   // h1 dead after conv2
    float* h4 = h2;   // h2 dead after conv3

    auto pack = [&](const float* src, _Float16* dst, int K, int Ncols, int tr) {
        int total = K * Ncols;
        pack_b_kernel<<<(total + 255) / 256, 256, 0, stream>>>(src, dst, K, Ncols, tr);
    };
    pack(l1_Wih, pL1ih, 64, 256, 1);   // B = Wih.T
    pack(l1_Whh, pL1hh, 64, 256, 1);
    pack(s1_Wself, pS1s, 64, 32, 0);
    pack(s1_Wnb,   pS1n, 64, 32, 0);
    pack(l2_Wih, pL2ih, 32, 128, 1);
    pack(l2_Whh, pL2hh, 32, 128, 1);
    pack(s2_Wself, pS2s, 32, 32, 0);
    pack(s2_Wnb,   pS2n, 32, 32, 0);
    pack(edge_W, pEW, 512, 32, 0);     // [16,32,32] flat == [512,32] row-major
    pack(edge_b, pEB, 32, 32, 0);

    const int tiles  = N / 16;
    const int blocks = tiles / WAVES;

    sage_lstm_kernel<64><<<blocks, WAVES * 32, 0, stream>>>(
        x, adj, pL1ih, pL1hh, l1_b, pS1s, pS1n, s1_b, h1, 1, N);
    sage_lstm_kernel<32><<<blocks, WAVES * 32, 0, stream>>>(
        h1, adj, pL2ih, pL2hh, l2_b, pS2s, pS2n, s2_b, h2, 1, N);
    nnconv_kernel<<<blocks, WAVES * 32, 0, stream>>>(
        h2, adj, efeat, pEW, pEB, nn_b, h3, N);
    sage_lstm_kernel<32><<<blocks, WAVES * 32, 0, stream>>>(
        h3, adj, pL2ih, pL2hh, l2_b, pS2s, pS2n, s2_b, h4, 0, N);
    pool_kernel<<<NG, N / NG, 0, stream>>>(h4, gate_W1, gate_b1, gate_W2, gate_b2, rp);
    final_mlp_kernel<<<1, NG, 0, stream>>>(rp, fin_W1, fin_b1, fin_W2, fin_b2,
                                           fin_W3, fin_b3, (float*)d_out);
}
